// LSTM_72885595013380
// MI455X (gfx1250) — matrix operand order repaired and checked by reading, not compiled, not run on hardware
//
#include <hip/hip_runtime.h>
#include <stdint.h>

// ---------------------------------------------------------------------------
// 2-layer LSTM for MI455X (gfx1250, wave32, WMMA).
// B=64 T=512 I=256 H=512 O=128.
//   * bulk f16 WMMA GEMMs for input projections, stored in a C-fragment-tiled
//     layout so the sequential scan seeds accumulators with single b128 loads
//   * persistent 32-block scan per layer: W_hh slice pinned in LDS via
//     GLOBAL_LOAD_ASYNC_TO_LDS_B128 (XOR-swizzled, bank-conflict-free reads),
//     cell state register-resident in WMMA C-fragment layout,
//     h ping-ponged through an L2-resident buffer + grid barrier per step.
// ---------------------------------------------------------------------------

typedef _Float16 h16;
typedef __attribute__((ext_vector_type(16))) _Float16 v16h;
typedef __attribute__((ext_vector_type(8)))  float    v8f;
typedef __attribute__((ext_vector_type(4)))  float    v4f;

#define B_  64
#define T_  512
#define I_  256
#define H_  512
#define G4_ 2048   // 4*H
#define O_  128
#define SCAN_BLOCKS 32

union FragU  { v4f f4[2]; v16h v; };
union Pack8H { v4f v; h16 h[8]; };

#define WMMA_F16(a, b, c) \
  __builtin_amdgcn_wmma_f32_16x16x32_f16(false, (a), false, (b), (short)0, (c), false, false)

// A fragment (16x32 f16, M x K). Lanes 0-15 hold row M=lane,
// K=k0..k0+15 contiguous; lanes 16-31 hold K=k0+16..k0+31.
static __device__ inline v16h load_a_frag(const h16* __restrict__ base, int ld, int k0) {
  const int lane = threadIdx.x & 31;
  const h16* p = base + (size_t)(lane & 15) * ld + k0 + ((lane >> 4) << 4);
  FragU u;
  u.f4[0] = *(const v4f*)(p);
  u.f4[1] = *(const v4f*)(p + 8);
  return u.v;
}

// B fragment (32x16 f16, K x N) from row-major W[n][k]: lane holds column
// N=lane&15 with 16 contiguous K values (low half K=k0.., high half K=k0+16..).
static __device__ inline v16h load_b_frag(const h16* __restrict__ base, int ld, int k0) {
  const int lane = threadIdx.x & 31;
  const h16* p = base + (size_t)(lane & 15) * ld + k0 + ((lane >> 4) << 4);
  FragU u;
  u.f4[0] = *(const v4f*)(p);
  u.f4[1] = *(const v4f*)(p + 8);
  return u.v;
}

// B fragment from the XOR-swizzled LDS W slice: [16 cols][512 K] halves,
// 8-half chunk c of column n stored at chunk (c ^ n) -> conflict-free
// ds_load_b128 across the 16 column-lanes.
static __device__ inline v16h load_b_frag_lds(const h16* base, int k0) {
  const int lane = threadIdx.x & 31;
  const int n    = lane & 15;
  const int kk   = k0 + ((lane >> 4) << 4);
  const int c0   = kk >> 3;
  const h16* row = base + n * 512;
  FragU u;
  u.f4[0] = *(const v4f*)(row + (((c0    ) ^ n) << 3));
  u.f4[1] = *(const v4f*)(row + (((c0 + 1) ^ n) << 3));
  return u.v;
}

// Async 16-byte copy global -> LDS (ASYNCcnt-tracked, no VGPR staging).
// The LDS destination is passed as a real pointer into the __shared__ array:
// the ptrtoint conversion *captures* it, so LLVM must treat the asm (with its
// memory clobber) as a legitimate writer of the LDS object -- otherwise the
// compiler proves the array is never stored and DCEs all ds_load_b128 reads.
// Generic LDS pointers carry the LDS byte address in their low 32 bits
// (aperture mapping).
static __device__ inline void async_copy_b128(const void* gsrc, h16* lds_dst) {
  uint32_t lds_addr = (uint32_t)(uintptr_t)lds_dst;
  asm volatile("global_load_async_to_lds_b128 %0, %1, off"
               :: "v"(lds_addr), "v"((uint64_t)(uintptr_t)gsrc)
               : "memory");
}
static __device__ inline void async_wait0() {
  asm volatile("s_wait_asynccnt 0x0" ::: "memory");
}

static __device__ inline float sigmoidf_(float x) { return 1.0f / (1.0f + __expf(-x)); }
static __device__ inline float tanhf_(float x) {
  float e = __expf(2.0f * x);
  return (e - 1.0f) / (e + 1.0f);
}

// ---------------------------------------------------------------------------
// Elementwise prep kernels
// ---------------------------------------------------------------------------

// x[b][t][i] fp32 -> xf16[t][b][i] f16  (scan-friendly [t][b] row ordering)
__global__ void cvt_x_kernel(const float* __restrict__ x, h16* __restrict__ xo, int n) {
  int i = blockIdx.x * blockDim.x + threadIdx.x;
  if (i >= n) return;
  int ii = i % I_;
  int t  = (i / I_) % T_;
  int b  = i / (I_ * T_);
  xo[((size_t)t * B_ + b) * I_ + ii] = (h16)x[i];
}

__global__ void cvt_f16_kernel(const float* __restrict__ in, h16* __restrict__ out, int n) {
  int i = blockIdx.x * blockDim.x + threadIdx.x;
  if (i < n) out[i] = (h16)in[i];
}

__global__ void bias_add_kernel(const float* __restrict__ a, const float* __restrict__ b,
                                float* __restrict__ o, int n) {
  int i = blockIdx.x * blockDim.x + threadIdx.x;
  if (i < n) o[i] = a[i] + b[i];
}

__global__ void scan_init_kernel(h16* __restrict__ hbuf0, int* __restrict__ cnt) {
  int i = blockIdx.x * blockDim.x + threadIdx.x;
  if (i < B_ * H_) hbuf0[i] = (h16)0.0f;
  if (i == 0) *cnt = 0;
}

// ---------------------------------------------------------------------------
// Bulk input-projection GEMM:  gates = A @ W^T + bias, stored f16 in a
// C-fragment-tiled layout:  G[t][gate(4)][ct(32)][col16(16)][row64(64)]
// (one GEMM 64-row tile == one timestep since B=64), so the scan can seed
// each accumulator with a single contiguous 16-byte load per lane.
// Block = 8 waves, block tile 64(M) x 128(N); wave tile 32x32.
// ---------------------------------------------------------------------------
__global__ __launch_bounds__(256) void gemm_gates_kernel(
    const h16* __restrict__ A, const h16* __restrict__ W,
    const float* __restrict__ bias, h16* __restrict__ G,
    int K) {
  const int wid  = threadIdx.x >> 5;
  const int wrow = wid & 1;
  const int wcol = wid >> 1;
  const int t  = blockIdx.x;                 // one 64-row tile per timestep
  const int m0 = t * 64 + wrow * 32;
  const int n0 = blockIdx.y * 128 + wcol * 32;

  v8f acc[2][2] = {};

  const h16* Arow0 = A + (size_t)m0 * K;
  const h16* Arow1 = A + (size_t)(m0 + 16) * K;
  const h16* Wrow0 = W + (size_t)n0 * K;
  const h16* Wrow1 = W + (size_t)(n0 + 16) * K;

  for (int k0 = 0; k0 < K; k0 += 32) {
    if (k0 + 64 < K) {  // stage next-next K tile (global_prefetch_b8)
      __builtin_prefetch(Arow0 + k0 + 64, 0, 1);
      __builtin_prefetch(Wrow0 + k0 + 64, 0, 1);
    }
    v16h a0 = load_a_frag(Arow0, K, k0);
    v16h a1 = load_a_frag(Arow1, K, k0);
    v16h b0 = load_b_frag(Wrow0, K, k0);
    v16h b1 = load_b_frag(Wrow1, K, k0);
    acc[0][0] = WMMA_F16(a0, b0, acc[0][0]);
    acc[0][1] = WMMA_F16(a0, b1, acc[0][1]);
    acc[1][0] = WMMA_F16(a1, b0, acc[1][0]);
    acc[1][1] = WMMA_F16(a1, b1, acc[1][1]);
  }

  const int lane = threadIdx.x & 31;
#pragma unroll
  for (int i = 0; i < 2; ++i) {
#pragma unroll
    for (int j = 0; j < 2; ++j) {
      const int col  = n0 + j * 16 + (lane & 15);
      const int g    = col >> 9;       // gate
      const int hcol = col & (H_ - 1); // column within gate
      const float bv = bias[col];
      const int brow = wrow * 32 + i * 16 + ((lane >> 4) << 3);  // row-in-t (0..63)
      Pack8H st;
#pragma unroll
      for (int e = 0; e < 8; ++e) st.h[e] = (h16)(acc[i][j][e] + bv);
      h16* dst = G + (((size_t)(t * 4 + g) * 32 + (hcol >> 4)) << 10)
                   + ((hcol & 15) << 6) + brow;
      *(v4f*)dst = st.v;  // one b128 store per accumulator
    }
  }
}

// ---------------------------------------------------------------------------
// Persistent recurrent scan.  32 blocks x 128 threads (4 waves).
// Block b owns h-columns [16*b, 16*b+16): its W_hh slice (4 gates x 16 cols x
// 512 K, 64 KB f16) is async-copied into LDS once; cell state lives in
// registers in WMMA C-fragment layout.  One grid barrier per timestep.
// ---------------------------------------------------------------------------
__global__ __launch_bounds__(128) void lstm_scan_kernel(
    const h16* __restrict__ G,        // tiled [T][4][32][16][64] f16 (bias folded)
    const h16* __restrict__ Whh,      // [2048][512]  f16
    h16*       __restrict__ hbuf,     // [2][B][512]  f16 ping-pong
    h16*       __restrict__ hseq_f16, // [T][B][512]  f16 (layer 0) or null
    float*     __restrict__ lstm_out, // [B][T][512]  fp32 (layer 1) or null
    float*     __restrict__ hT,       // [B][512] fp32
    float*     __restrict__ cT,       // [B][512] fp32
    int*       __restrict__ barrier_cnt) {
  __shared__ h16 Wl[4 * 16 * 512];  // 64 KB, XOR-swizzled

  const int tid  = threadIdx.x;
  const int wid  = tid >> 5;
  const int lane = tid & 31;
  const int ct   = blockIdx.x;             // owned h-column tile
  const int c0   = ct * 16;                // owned h-column base
  const int m0   = wid * 16;               // batch-row tile of this wave

  // ---- Async preload of the W_hh slice straight into LDS (no VGPR staging).
  // 4096 16B chunks / 128 threads = 32 issues per thread, then one wait.
#pragma unroll 4
  for (int it = 0; it < (4 * 16 * 64) / 128; ++it) {
    int id = it * 128 + tid;
    int ck = id & 63;           // 8-half chunk along K
    int n  = (id >> 6) & 15;    // local column
    int g  = id >> 10;          // gate
    const h16* src = Whh + ((size_t)(g * 512 + c0 + n) * 512) + ck * 8;
    h16*       dst = Wl + ((g * 16 + n) * 512 + ((ck ^ n) << 3));
    async_copy_b128(src, dst);
  }
  async_wait0();
  __syncthreads();

  const int coln  = c0 + (lane & 15);        // this lane's h-column
  const int rbase = m0 + ((lane >> 4) << 3); // first batch row of C elements

  v8f c_acc = {};   // cell state, register-resident across all 512 steps
  int target = 0;

  for (int t = 0; t < T_; ++t) {
    const h16* hprev = hbuf + ((t & 1) ? (B_ * H_) : 0);
    h16*       hnext = hbuf + ((t & 1) ? 0 : (B_ * H_));

    // Seed accumulators from tiled G: one contiguous b128 load per gate.
    v8f acc[4];
#pragma unroll
    for (int g = 0; g < 4; ++g) {
      Pack8H ld;
      ld.v = *(const v4f*)(G + (((size_t)(t * 4 + g) * 32 + ct) << 10)
                             + ((lane & 15) << 6) + rbase);
      v8f a;
#pragma unroll
      for (int e = 0; e < 8; ++e) a[e] = (float)ld.h[e];
      acc[g] = a;
    }

    // gates += h_prev @ W_hh^T   (K = 512 -> 16 wmma per gate)
    const h16* Ab = hprev + (size_t)m0 * H_;
#pragma unroll 4
    for (int k0 = 0; k0 < H_; k0 += 32) {
      v16h af = load_a_frag(Ab, H_, k0);
#pragma unroll
      for (int g = 0; g < 4; ++g) {
        v16h bf = load_b_frag_lds(Wl + g * 16 * 512, k0);
        acc[g] = WMMA_F16(af, bf, acc[g]);
      }
    }

    // Elementwise LSTM cell (register-aligned with C-fragment layout).
    v8f hnew;
#pragma unroll
    for (int e = 0; e < 8; ++e) {
      float ig = sigmoidf_(acc[0][e]);
      float fg = sigmoidf_(acc[1][e]);
      float gg = tanhf_(acc[2][e]);
      float og = sigmoidf_(acc[3][e]);
      float c  = fg * c_acc[e] + ig * gg;
      c_acc[e] = c;
      hnew[e]  = og * tanhf_(c);
    }

    // Publish h for the next step + sequence outputs.
#pragma unroll
    for (int e = 0; e < 8; ++e) {
      const int row = rbase + e;  // batch index
      const h16 hv = (h16)hnew[e];
      hnext[(size_t)row * H_ + coln] = hv;
      if (hseq_f16) hseq_f16[((size_t)t * B_ + row) * H_ + coln] = hv;
      if (lstm_out) lstm_out[((size_t)row * T_ + t) * H_ + coln] = hnew[e];
    }
    if (t == T_ - 1) {
#pragma unroll
      for (int e = 0; e < 8; ++e) {
        const int row = rbase + e;
        hT[(size_t)row * H_ + coln] = hnew[e];
        cT[(size_t)row * H_ + coln] = c_acc[e];
      }
    }

    // Grid barrier: device-scope release of hnext, arrive, acquire-spin.
    target += SCAN_BLOCKS;
    if (t < T_ - 1) {
      __threadfence();
      __syncthreads();
      if (tid == 0) {
        __hip_atomic_fetch_add(barrier_cnt, 1, __ATOMIC_RELEASE, __HIP_MEMORY_SCOPE_AGENT);
        while (__hip_atomic_load(barrier_cnt, __ATOMIC_ACQUIRE, __HIP_MEMORY_SCOPE_AGENT) < target) {
          __builtin_amdgcn_s_sleep(1);
        }
      }
      __syncthreads();
    }
  }
}

// ---------------------------------------------------------------------------
// out = h_last @ W_out^T + b_out   (64 x 128, K = 512).  One block, 8 waves.
// ---------------------------------------------------------------------------
__global__ __launch_bounds__(256) void out_proj_kernel(
    const h16* __restrict__ hlast,  // [64][512] f16
    const h16* __restrict__ Wout,   // [128][512] f16
    const float* __restrict__ bout, // [128]
    float* __restrict__ out) {      // [64][128]
  const int wid = threadIdx.x >> 5;
  const int m0  = (wid >> 1) * 16;   // 4 row tiles
  const int n0  = (wid & 1) * 64;    // 2 column halves of 64

  v8f acc[4] = {};
  const h16* Ab = hlast + (size_t)m0 * H_;
  for (int k0 = 0; k0 < H_; k0 += 32) {
    v16h a = load_a_frag(Ab, H_, k0);
#pragma unroll
    for (int j = 0; j < 4; ++j) {
      v16h b = load_b_frag(Wout + (size_t)(n0 + j * 16) * H_, H_, k0);
      acc[j] = WMMA_F16(a, b, acc[j]);
    }
  }
  const int lane = threadIdx.x & 31;
#pragma unroll
  for (int j = 0; j < 4; ++j) {
    const int col   = n0 + j * 16 + (lane & 15);
    const float bv  = bout[col];
    const int rbase = m0 + ((lane >> 4) << 3);
#pragma unroll
    for (int e = 0; e < 8; ++e) out[(size_t)(rbase + e) * O_ + col] = acc[j][e] + bv;
  }
}

// ---------------------------------------------------------------------------
// Host side
// ---------------------------------------------------------------------------
extern "C" void kernel_launch(void* const* d_in, const int* in_sizes, int n_in,
                              void* d_out, int out_size, void* d_ws, size_t ws_size,
                              hipStream_t stream) {
  (void)in_sizes; (void)n_in; (void)out_size; (void)ws_size;
  const float* x    = (const float*)d_in[0];
  const float* Wih0 = (const float*)d_in[1];
  const float* Whh0 = (const float*)d_in[2];
  const float* bih0 = (const float*)d_in[3];
  const float* bhh0 = (const float*)d_in[4];
  const float* Wih1 = (const float*)d_in[5];
  const float* Whh1 = (const float*)d_in[6];
  const float* bih1 = (const float*)d_in[7];
  const float* bhh1 = (const float*)d_in[8];
  const float* Wout = (const float*)d_in[9];
  const float* bout = (const float*)d_in[10];
  float* out = (float*)d_out;

  // Workspace carve-up (~192 MB, 256B-aligned slices).
  char* ws = (char*)d_ws;
  size_t off = 0;
  auto alloc = [&](size_t bytes) -> void* {
    void* p = ws + off;
    off = (off + bytes + 255) & ~(size_t)255;
    return p;
  };
  h16*   xf    = (h16*)alloc(sizeof(h16) * (size_t)T_ * B_ * I_);
  h16*   hs0   = (h16*)alloc(sizeof(h16) * (size_t)T_ * B_ * H_);
  h16*   G     = (h16*)alloc(sizeof(h16) * (size_t)T_ * B_ * G4_);
  h16*   wih0  = (h16*)alloc(sizeof(h16) * (size_t)G4_ * I_);
  h16*   whh0  = (h16*)alloc(sizeof(h16) * (size_t)G4_ * H_);
  h16*   wih1  = (h16*)alloc(sizeof(h16) * (size_t)G4_ * H_);
  h16*   whh1  = (h16*)alloc(sizeof(h16) * (size_t)G4_ * H_);
  h16*   wout  = (h16*)alloc(sizeof(h16) * (size_t)O_ * H_);
  float* bias0 = (float*)alloc(sizeof(float) * G4_);
  float* bias1 = (float*)alloc(sizeof(float) * G4_);
  h16*   hbuf  = (h16*)alloc(sizeof(h16) * 2 * B_ * H_);
  int*   cnt   = (int*)alloc(256);

  // d_out layout: out | lstm_out | hidden | cell
  float* out_dst  = out;
  float* lstm_out = out + (size_t)B_ * O_;
  float* hidden   = lstm_out + (size_t)B_ * T_ * H_;
  float* cell     = hidden + (size_t)2 * B_ * H_;

  // --- conversions ---
  {
    int n = B_ * T_ * I_;
    cvt_x_kernel<<<(n + 255) / 256, 256, 0, stream>>>(x, xf, n);
  }
  cvt_f16_kernel<<<(G4_ * I_ + 255) / 256, 256, 0, stream>>>(Wih0, wih0, G4_ * I_);
  cvt_f16_kernel<<<(G4_ * H_ + 255) / 256, 256, 0, stream>>>(Whh0, whh0, G4_ * H_);
  cvt_f16_kernel<<<(G4_ * H_ + 255) / 256, 256, 0, stream>>>(Wih1, wih1, G4_ * H_);
  cvt_f16_kernel<<<(G4_ * H_ + 255) / 256, 256, 0, stream>>>(Whh1, whh1, G4_ * H_);
  cvt_f16_kernel<<<(O_ * H_ + 255) / 256, 256, 0, stream>>>(Wout, wout, O_ * H_);
  bias_add_kernel<<<(G4_ + 255) / 256, 256, 0, stream>>>(bih0, bhh0, bias0, G4_);
  bias_add_kernel<<<(G4_ + 255) / 256, 256, 0, stream>>>(bih1, bhh1, bias1, G4_);

  const int M = T_ * B_;  // 32768

  // --- layer 0 ---
  gemm_gates_kernel<<<dim3(M / 64, G4_ / 128), 256, 0, stream>>>(xf, wih0, bias0, G, I_);
  scan_init_kernel<<<(B_ * H_ + 255) / 256, 256, 0, stream>>>(hbuf, cnt);
  lstm_scan_kernel<<<SCAN_BLOCKS, 128, 0, stream>>>(G, whh0, hbuf, hs0, nullptr,
                                                    hidden, cell, cnt);

  // --- layer 1 ---
  gemm_gates_kernel<<<dim3(M / 64, G4_ / 128), 256, 0, stream>>>(hs0, wih1, bias1, G, H_);
  scan_init_kernel<<<(B_ * H_ + 255) / 256, 256, 0, stream>>>(hbuf, cnt);
  lstm_scan_kernel<<<SCAN_BLOCKS, 128, 0, stream>>>(G, whh1, hbuf, nullptr, lstm_out,
                                                    hidden + (size_t)B_ * H_,
                                                    cell + (size_t)B_ * H_, cnt);

  // --- output projection (T even -> final h sits in hbuf[0]) ---
  out_proj_kernel<<<1, 256, 0, stream>>>(hbuf, wout, bout, out_dst);
}
